// MemoryLayer_41747082117534
// MI455X (gfx1250) — matrix operand order repaired and checked by reading, not compiled, not run on hardware
//
#include <hip/hip_runtime.h>

// ---------------------------------------------------------------------------
// MemoryLayer (hetero-graph relational message passing) for MI455X / gfx1250
//   msg[e,o] = sum_m coef[e,m] * (h_src[e,:] @ Ww[r][o,:,m])
//   coef     = LeakyReLU(h_dst @ Wc[r]^T + bc[r], 0.2)
//   out[n]   = mean_{e: dst(e)=n} msg[e] + h_bias + self_loop_term(feat[n])
// Strategy: bin edges by relation; stage W2[r]=[64,1024] f16 in LDS (128 KB)
// via async global->LDS copies; 16-edge tiles through v_wmma_f32_16x16x32_f16
// (130 WMMA-equivalents per tile, f32 accumulate); rank-1 coef weighting in
// VALU with ds_bpermute-style shuffles; scatter-add with f32 atomics.
// ---------------------------------------------------------------------------

typedef __attribute__((ext_vector_type(16))) _Float16 v16h;
typedef __attribute__((ext_vector_type(8)))  float    v8f;

#define NN   20000
#define EE   320000
#define RR   4
#define DIN  64
#define DOUT 64
#define DMEM 16
#define NEG  0.2f

static __device__ __forceinline__ v8f wmma16(v16h a, v16h b, v8f c) {
  // D = A(16x32 f16) x B(32x16 f16) + C(16x16 f32)
  return __builtin_amdgcn_wmma_f32_16x16x32_f16(
      /*neg_a=*/false, a, /*neg_b=*/false, b,
      /*c_mod=*/(short)0, c, /*reuse_a=*/false, /*reuse_b=*/false);
}

// Build A-operand (16 edges x 32 features, f16) for one K-step directly from a
// gathered f32 feature row. Per ISA 16-bit A layout:
//   lane l, half h:  M = l%16,  K = (h%8) + 8*(2*(h>=8) + (l>=16))
// => per lane two runs of 8 consecutive floats at offsets hiK and 16+hiK.
static __device__ __forceinline__ v16h build_a(const float* __restrict__ row,
                                               int kstep, int lane) {
  const int hiK = (lane & 16) >> 1;                 // 0 or 8
  const float* p = row + kstep * 32 + hiK;
  float4 f0 = *(const float4*)(p + 0);
  float4 f1 = *(const float4*)(p + 4);
  float4 f2 = *(const float4*)(p + 16);
  float4 f3 = *(const float4*)(p + 20);
  v16h a;
  a[0]  = (_Float16)f0.x; a[1]  = (_Float16)f0.y;
  a[2]  = (_Float16)f0.z; a[3]  = (_Float16)f0.w;
  a[4]  = (_Float16)f1.x; a[5]  = (_Float16)f1.y;
  a[6]  = (_Float16)f1.z; a[7]  = (_Float16)f1.w;
  a[8]  = (_Float16)f2.x; a[9]  = (_Float16)f2.y;
  a[10] = (_Float16)f2.z; a[11] = (_Float16)f2.w;
  a[12] = (_Float16)f3.x; a[13] = (_Float16)f3.y;
  a[14] = (_Float16)f3.z; a[15] = (_Float16)f3.w;
  return a;
}

// ----------------------------- small helpers --------------------------------

__global__ void k_zero(float* agg, float* deg, int* counters) {
  int i = blockIdx.x * blockDim.x + threadIdx.x;
  if (i < NN * DOUT) agg[i] = 0.f;
  if (i < NN)        deg[i] = 0.f;
  if (i < 16)        counters[i] = 0;
}

// Pre-pack Ww into B-operand order, f16.
// Tile t = m*8 + ob*2 + ks holds B[k][n] = Ww[rel][ob*16+n][ks*32+k][m]
// B layout: lane l holds column n=l%16, halves h -> K = (h%8) + 8*(l>=16) + 16*(h>=8).
// Buffer index: rel*65536 + t*512 + lane*16 + h  (contiguous 32B per lane).
__global__ void k_prep_w2(const float* __restrict__ Ww,
                          const float* __restrict__ Wwn,
                          _Float16* __restrict__ W2B) {
  int idx = blockIdx.x * blockDim.x + threadIdx.x;
  if (idx >= (RR + 1) * 65536) return;
  int rel  = idx >> 16;
  int rem  = idx & 65535;
  int tile = rem >> 9;
  int lh   = rem & 511;
  int lane = lh >> 4;
  int h    = lh & 15;
  int m    = tile >> 3;
  int ob   = (tile >> 1) & 3;
  int ks   = tile & 1;
  int K    = ks * 32 + (h & 7) + ((lane & 16) >> 1) + ((h & 8) << 1);
  int o    = ob * 16 + (lane & 15);
  float v;
  if (rel < RR) v = Ww[(((size_t)rel * DOUT + o) * DIN + K) * DMEM + m];
  else          v = Wwn[((size_t)o * DIN + K) * DMEM + m];
  W2B[idx] = (_Float16)v;
}

// Pre-pack Wc^T into B-operand order (B[k][m] = Wc[rel][m][K]) and bc.
__global__ void k_prep_wc(const float* __restrict__ Wc,
                          const float* __restrict__ Wcn,
                          const float* __restrict__ bc,
                          const float* __restrict__ bcn,
                          _Float16* __restrict__ WcB,
                          float* __restrict__ bcB) {
  int idx = blockIdx.x * blockDim.x + threadIdx.x;
  if (idx < (RR + 1) * 1024) {
    int rel  = idx >> 10;
    int rem  = idx & 1023;
    int ks   = rem >> 9;
    int lh   = rem & 511;
    int lane = lh >> 4;
    int h    = lh & 15;
    int K    = ks * 32 + (h & 7) + ((lane & 16) >> 1) + ((h & 8) << 1);
    int m    = lane & 15;
    float v  = (rel < RR) ? Wc[((size_t)rel * DMEM + m) * DIN + K]
                          : Wcn[(size_t)m * DIN + K];
    WcB[idx] = (_Float16)v;
  } else if (idx < (RR + 1) * 1024 + (RR + 1) * 16) {
    int j   = idx - (RR + 1) * 1024;
    int rel = j >> 4;
    int m   = j & 15;
    bcB[j] = (rel < RR) ? bc[rel * 16 + m] : bcn[m];
  }
}

__global__ void k_count(const int* __restrict__ ety, int* counters, int n) {
  int i = blockIdx.x * blockDim.x + threadIdx.x;
  if (i < n) atomicAdd(&counters[ety[i]], 1);
}

__global__ void k_offsets(int* c) {
  if (threadIdx.x == 0 && blockIdx.x == 0) {
    int acc = 0;
    for (int r = 0; r < RR; ++r) { c[4 + r] = acc; c[8 + r] = acc; acc += c[r]; }
  }
}

__global__ void k_scatter(const int* __restrict__ ety, int* counters,
                          int* __restrict__ bins, int n) {
  int i = blockIdx.x * blockDim.x + threadIdx.x;
  if (i < n) {
    int r = ety[i];
    int pos = atomicAdd(&counters[8 + r], 1);
    bins[pos] = i;
  }
}

__global__ void k_deg(const int* __restrict__ dst, float* __restrict__ deg, int n) {
  int i = blockIdx.x * blockDim.x + threadIdx.x;
  if (i < n) atomicAdd(&deg[dst[i]], 1.0f);
}

__global__ void k_final(const float* __restrict__ agg,
                        const float* __restrict__ deg,
                        const float* __restrict__ hb,
                        float* __restrict__ out) {
  int i = blockIdx.x * blockDim.x + threadIdx.x;
  if (i < NN * DOUT) {
    float d = deg[i >> 6];
    d = d > 1.f ? d : 1.f;
    out[i] = agg[i] / d + hb[i & 63];
  }
}

// ------------------------------ main GEMM -----------------------------------
// Edge mode (selfMode=0): relation = blockIdx.y, tiles from its edge bin,
//   atomic scatter-add of msg into agg[dst].
// Self mode (selfMode=1): relation slot 4 (node hypernet), identity indices,
//   direct += into out (each element has a unique writer).
__global__ __launch_bounds__(128) void k_gemm(
    const float* __restrict__ feat,
    const int* __restrict__ src, const int* __restrict__ dst,
    const int* __restrict__ bins, const int* __restrict__ counters,
    const _Float16* __restrict__ W2B, const _Float16* __restrict__ WcB,
    const float* __restrict__ bcB,
    float* __restrict__ outp, int selfMode) {
  extern __shared__ _Float16 sW2[];   // 64K halves = 128 KB (this relation's W2)
  const int rel = selfMode ? RR : blockIdx.y;

  { // async global->LDS fill (ASYNCcnt-tracked, no data VGPRs): 8192 x 16B
    const char* g = (const char*)(W2B + (size_t)rel * 65536);
    const unsigned ldsBase = (unsigned)(uintptr_t)sW2;  // LDS offset = addr[31:0]
    for (int i = threadIdx.x; i < 8192; i += blockDim.x) {
      unsigned long long ga = (unsigned long long)(g + (size_t)i * 16);
      unsigned la = ldsBase + (unsigned)i * 16u;
      asm volatile("global_load_async_to_lds_b128 %0, %1, off"
                   :: "v"(la), "v"(ga) : "memory");
    }
    asm volatile("s_wait_asynccnt 0x0" ::: "memory");
  }
  __syncthreads();

  const int lane = threadIdx.x & 31;
  const int wave = threadIdx.x >> 5;
  const int hi   = (lane >> 4) & 1;
  const int col  = lane & 15;

  const int cnt     = selfMode ? NN : counters[rel];
  const int nTiles  = (cnt + 15) >> 4;
  const int binBase = selfMode ? 0 : counters[4 + rel];
  const int nw      = blockDim.x >> 5;
  const int wgid    = blockIdx.x * nw + wave;
  const int wstride = gridDim.x * nw;

  const float bcv = bcB[rel * 16 + col];
  const _Float16* wcp = WcB + ((size_t)rel * 2) * 512 + lane * 16;
  const _Float16* sW2lane = sW2 + lane * 16;

  for (int t = wgid; t < nTiles; t += wstride) {
    const int tbase = t << 4;
    int p = tbase + col;
    if (p >= cnt) p = cnt - 1;          // clamp padding lanes (masked at store)
    int sN, dN;
    if (selfMode) { sN = p; dN = p; }
    else { int eId = bins[binBase + p]; sN = src[eId]; dN = dst[eId]; }
    const float* srow = feat + (size_t)sN * DIN;
    const float* drow = feat + (size_t)dN * DIN;

    // coef = LeakyReLU(H_dst @ Wc^T + bc)   [16 edges x 16 mem]
    // (Wc B-operands reloaded per tile from L2 to keep live ranges short.)
    v8f cc = {};
    cc = wmma16(build_a(drow, 0, lane), *(const v16h*)(wcp), cc);
    cc = wmma16(build_a(drow, 1, lane), *(const v16h*)(wcp + 512), cc);
#pragma unroll
    for (int v = 0; v < 8; ++v) {
      float x = cc[v] + bcv;
      cc[v] = x > 0.f ? x : NEG * x;
    }

    const v16h as0 = build_a(srow, 0, lane);
    const v16h as1 = build_a(srow, 1, lane);

    v8f acc[4] = {};                    // msg tile: 16 edges x 64 out (f32)
#pragma unroll 4
    for (int m = 0; m < 16; ++m) {
      // broadcast coef[e, m] to all lanes of row e (WMMA C layout aware)
      float cf[8];
#pragma unroll
      for (int v = 0; v < 8; ++v) cf[v] = __shfl(cc[v], m + (lane & 16), 32);
      const _Float16* tb = sW2lane + m * (8 * 512);
#pragma unroll
      for (int ob = 0; ob < 4; ++ob) {
        v8f tt = {};
        tt = wmma16(as0, *(const v16h*)(tb + (ob * 2 + 0) * 512), tt);
        tt = wmma16(as1, *(const v16h*)(tb + (ob * 2 + 1) * 512), tt);
#pragma unroll
        for (int v = 0; v < 8; ++v) acc[ob][v] += cf[v] * tt[v];
      }
    }

    // scatter msg: element (e = v + 8*hi, o = ob*16 + col)
#pragma unroll
    for (int v = 0; v < 8; ++v) {
      const int e  = v + hi * 8;
      const int pe = tbase + e;
      if (pe < cnt) {
        int dn = selfMode ? pe : __shfl(dN, e, 32);
        float* base = outp + (size_t)dn * DOUT + col;
        if (selfMode) {
          base[0]  += acc[0][v];
          base[16] += acc[1][v];
          base[32] += acc[2][v];
          base[48] += acc[3][v];
        } else {
          atomicAdd(base + 0,  acc[0][v]);
          atomicAdd(base + 16, acc[1][v]);
          atomicAdd(base + 32, acc[2][v]);
          atomicAdd(base + 48, acc[3][v]);
        }
      }
    }
  }
}

// ------------------------------- launcher -----------------------------------

extern "C" void kernel_launch(void* const* d_in, const int* in_sizes, int n_in,
                              void* d_out, int out_size, void* d_ws, size_t ws_size,
                              hipStream_t stream) {
  (void)in_sizes; (void)n_in; (void)out_size; (void)ws_size;
  const float* feat = (const float*)d_in[0];
  const int*   srcp = (const int*)d_in[1];
  const int*   dstp = (const int*)d_in[2];
  const int*   ety  = (const int*)d_in[3];
  const float* Wc   = (const float*)d_in[4];
  const float* bc   = (const float*)d_in[5];
  const float* Ww   = (const float*)d_in[6];
  const float* Wcn  = (const float*)d_in[7];
  const float* bcn  = (const float*)d_in[8];
  const float* Wwn  = (const float*)d_in[9];
  const float* hb   = (const float*)d_in[10];
  float* out = (float*)d_out;

  // workspace layout
  char* ws = (char*)d_ws;
  int*      counters = (int*)ws;                        // 16 ints
  int*      bins     = (int*)(ws + 64);                 // EE ints
  size_t off = 64 + (size_t)EE * 4;
  float*    deg = (float*)(ws + off); off += (size_t)NN * 4;
  float*    agg = (float*)(ws + off); off += (size_t)NN * DOUT * 4;
  _Float16* W2B = (_Float16*)(ws + off); off += (size_t)(RR + 1) * 65536 * 2;
  _Float16* WcB = (_Float16*)(ws + off); off += (size_t)(RR + 1) * 1024 * 2;
  float*    bcB = (float*)(ws + off);

  const int LDS_BYTES = 65536 * 2;  // 128 KB dynamic LDS (CDNA5: 320 KB/WGP)
  hipFuncSetAttribute(reinterpret_cast<const void*>(&k_gemm),
                      hipFuncAttributeMaxDynamicSharedMemorySize, LDS_BYTES);

  k_zero<<<(NN * DOUT + 255) / 256, 256, 0, stream>>>(agg, deg, counters);
  k_prep_w2<<<((RR + 1) * 65536 + 255) / 256, 256, 0, stream>>>(Ww, Wwn, W2B);
  k_prep_wc<<<((RR + 1) * 1024 + (RR + 1) * 16 + 255) / 256, 256, 0, stream>>>(
      Wc, Wcn, bc, bcn, WcB, bcB);
  k_count<<<(EE + 255) / 256, 256, 0, stream>>>(ety, counters, EE);
  k_offsets<<<1, 32, 0, stream>>>(counters);
  k_scatter<<<(EE + 255) / 256, 256, 0, stream>>>(ety, counters, bins, EE);
  k_deg<<<(EE + 255) / 256, 256, 0, stream>>>(dstp, deg, EE);

  // edge-message GEMMs: one relation per blockIdx.y, persistent waves
  dim3 g(64, RR);
  k_gemm<<<g, 128, LDS_BYTES, stream>>>(feat, srcp, dstp, bins, counters,
                                        W2B, WcB, bcB, agg, 0);
  // mean + bias
  k_final<<<(NN * DOUT + 255) / 256, 256, 0, stream>>>(agg, deg, hb, out);
  // self-loop term added directly into out
  k_gemm<<<dim3(80, 1), 128, LDS_BYTES, stream>>>(feat, nullptr, nullptr, nullptr,
                                                  counters, W2B, WcB, bcB, out, 1);
}
